// PWCFlow_58188216926852
// MI455X (gfx1250) — matrix operand order repaired and checked by reading, not compiled, not run on hardware
//
#include <hip/hip_runtime.h>
#include <hip/hip_bf16.h>
#include <math.h>

typedef __attribute__((ext_vector_type(16))) _Float16 v16h;
typedef __attribute__((ext_vector_type(8)))  float    v8f;

static __device__ __forceinline__ int iclamp(int v, int lo, int hi) {
    return v < lo ? lo : (v > hi ? hi : v);
}

// ---------------------------------------------------------------------------
// Weight pre-pack: HWIO f32 [3][3][Cin][Cout]  ->  f16 [CoutPad][9*CinPad]
// laid out so a B fragment for (n, kchunk) is 16 consecutive halves (32B,
// aligned). Zero-padded in both Cin (per tap) and Cout.
// ---------------------------------------------------------------------------
__global__ void pack_weights_kernel(const float* __restrict__ w,
                                    int Cin, int Cout,
                                    _Float16* __restrict__ dst,
                                    int CinPad, int CoutPad)
{
    const int KtotPad = 9 * CinPad;
    const int N = CoutPad * KtotPad;
    const int idx = blockIdx.x * blockDim.x + threadIdx.x;
    if (idx >= N) return;
    const int n  = idx / KtotPad;
    const int k  = idx - n * KtotPad;
    const int tap = k / CinPad;
    const int ci  = k - tap * CinPad;
    float v = 0.0f;
    if (n < Cout && ci < Cin)
        v = w[((size_t)(tap * Cin + ci)) * Cout + n];
    dst[idx] = (_Float16)v;
}

// ---------------------------------------------------------------------------
// Generic 3x3 conv (SAME, dilation), implicit GEMM on v_wmma_f32_16x16x32_f16.
// One wave -> 16 pixels (M) x NT*16 output channels; NT accumulators share one
// A fragment (4x A reuse at NT=4). The Cin-chunk loop is an explicit two-stage
// ping-pong: WMMAs consume buffer X while the next chunk loads into buffer Y,
// with no register-copy hand-off. Tap loops stay rolled for code size.
// Tail/border A loads are branchless (safe clamped pointer + v_cndmask).
// ---------------------------------------------------------------------------
template <int NT>
__global__ void __launch_bounds__(128)
conv3x3_wmma_kernel(const float* __restrict__ in, int inStride, int inOff, int Cin,
                    const _Float16* __restrict__ wpack, int CinPad,
                    const float* __restrict__ bias,
                    float* __restrict__ out, int outStride, int outOff, int Cout,
                    const float* __restrict__ resid, int residStride,
                    int B, int H, int W, int dil, int leaky)
{
    const int P       = B * H * W;
    const int Mtiles  = (P + 15) >> 4;
    const int NGroups = (Cout + NT * 16 - 1) / (NT * 16);
    const int wid     = blockIdx.x * (blockDim.x >> 5) + (threadIdx.x >> 5);
    if (wid >= Mtiles * NGroups) return;

    const int mt   = wid / NGroups;
    const int ng   = wid - mt * NGroups;
    const int lane = threadIdx.x & 31;
    const int m    = lane & 15;         // A row / D column-lane
    const int hi   = lane >> 4;         // half-wave selector
    const int n0   = ng * (NT * 16) + m;

    const int  pm       = mt * 16 + m;
    const bool rowValid = pm < P;
    int px = 0, py = 0, pb = 0;
    if (rowValid) {
        px = pm % W;
        int t = pm / W;
        py = t % H;
        pb = t / H;
    }

    const int KtotPad = 9 * CinPad;
    const _Float16* wp0 = wpack + (size_t)n0 * KtotPad + hi * 16;

    v8f acc[NT] = {};

    int kbase = 0;
    #pragma clang loop unroll(disable)
    for (int kh = 0; kh < 3; ++kh) {
        const int yy = py + (kh - 1) * dil;
        #pragma clang loop unroll(disable)
        for (int kw = 0; kw < 3; ++kw) {
            const int xx = px + (kw - 1) * dil;
            const bool tapValid =
                rowValid && yy >= 0 && yy < H && xx >= 0 && xx < W;
            // Safe base: for invalid taps point at `in` and mask to zero.
            const float* sbase = tapValid
                ? in + (((size_t)pb * H + yy) * W + xx) * inStride + inOff
                : in;
            const _Float16* wtap = wp0 + kbase;
            __builtin_prefetch(wtap, 0, 1);   // global_prefetch_b8

            // ---- chunk loader: A (16x32 f16, contiguous k per half-wave) + NT B fragments ---
            // lanes 0-15: K = {0..15}; lanes 16-31: K = {16..31}
            auto loadChunk = [&](int ci0, v16h& af, v16h (&bf)[NT]) {
                if (tapValid && ci0 + 32 <= Cin) {
                    const float* bp = sbase + ci0 + hi * 16;
                    #pragma unroll
                    for (int i = 0; i < 8; ++i)
                        af[i] = (_Float16)bp[i];
                    #pragma unroll
                    for (int i = 8; i < 16; ++i)
                        af[i] = (_Float16)bp[i];
                } else {
                    #pragma unroll
                    for (int i = 0; i < 16; ++i) {
                        const int ci = ci0 + i + hi * 16;
                        const float v = sbase[ci < Cin ? ci : 0]; // always safe
                        af[i] = (tapValid && ci < Cin) ? (_Float16)v
                                                       : (_Float16)0.0f;
                    }
                }
                #pragma unroll
                for (int t = 0; t < NT; ++t)
                    bf[t] = *(const v16h*)(wtap + (size_t)t * 16 * KtotPad + ci0);
            };

            auto mma = [&](const v16h& af, const v16h (&bf)[NT]) {
                #pragma unroll
                for (int t = 0; t < NT; ++t)
                    acc[t] = __builtin_amdgcn_wmma_f32_16x16x32_f16(
                        /*neg_a=*/false, af, /*neg_b=*/false, bf[t],
                        /*c_mod=*/(short)0, acc[t],
                        /*reuse_a=*/false, /*reuse_b=*/false);
            };

            // Two-stage software pipeline (ping-pong, no hand-off copies).
            v16h a0, a1;
            v16h b0[NT], b1[NT];
            loadChunk(0, a0, b0);
            int ci0 = 0;
            for (;;) {
                int nci = ci0 + 32;
                if (nci < Cin) loadChunk(nci, a1, b1);  // issue loads first
                mma(a0, b0);
                ci0 = nci;
                if (ci0 >= Cin) break;

                nci = ci0 + 32;
                if (nci < Cin) loadChunk(nci, a0, b0);
                mma(a1, b1);
                ci0 = nci;
                if (ci0 >= Cin) break;
            }
            kbase += CinPad;
        }
    }

    #pragma unroll
    for (int t = 0; t < NT; ++t) {
        const int n = n0 + 16 * t;
        if (n < Cout) {
            const float bs = bias[n];
            #pragma unroll
            for (int r = 0; r < 8; ++r) {
                const int p = mt * 16 + r + hi * 8;   // C/D: VGPR r -> M=r+8*hi
                if (p < P) {
                    float v = acc[t][r] + bs;
                    if (leaky) v = v > 0.0f ? v : 0.1f * v;
                    if (resid) v += resid[(size_t)p * residStride + n];
                    out[(size_t)p * outStride + outOff + n] = v;
                }
            }
        }
    }
}

// ---------------------------------------------------------------------------
// Bilinear warp: dst = resample(src, grid + flow), zero outside.
// ---------------------------------------------------------------------------
__global__ void warp_kernel(const float* __restrict__ src,
                            const float* __restrict__ flow,
                            float* __restrict__ dst,
                            int B, int H, int W, int C)
{
    const int P = B * H * W;
    const int p = blockIdx.x * blockDim.x + threadIdx.x;
    if (p >= P) return;
    const int x = p % W;
    int t = p / W;
    const int y = t % H;
    const int b = t / H;

    const float cy = (float)y + flow[(size_t)p * 2 + 0];
    const float cx = (float)x + flow[(size_t)p * 2 + 1];
    const float y0f = floorf(cy), x0f = floorf(cx);
    const float wy = cy - y0f, wx = cx - x0f;
    const int y0 = (int)y0f, x0 = (int)x0f;

    for (int c = 0; c < C; ++c) {
        float o = 0.0f;
        #pragma unroll
        for (int dy = 0; dy < 2; ++dy)
            #pragma unroll
            for (int dx = 0; dx < 2; ++dx) {
                const int yi = y0 + dy, xi = x0 + dx;
                if (yi >= 0 && yi < H && xi >= 0 && xi < W) {
                    const float wgt = (dy ? wy : 1.0f - wy) * (dx ? wx : 1.0f - wx);
                    o += wgt * src[(((size_t)b * H + yi) * W + xi) * C + c];
                }
            }
        dst[(size_t)p * C + c] = o;
    }
}

// ---------------------------------------------------------------------------
// Per-image sum / sum-of-squares reduction (for _normalize_pair moments).
// ---------------------------------------------------------------------------
__global__ void moments_kernel(const float* __restrict__ t, int perImage,
                               float* __restrict__ acc, int accOff)
{
    const int b = blockIdx.y;
    float s = 0.0f, s2 = 0.0f;
    for (int i = blockIdx.x * blockDim.x + threadIdx.x; i < perImage;
         i += gridDim.x * blockDim.x) {
        const float v = t[(size_t)b * perImage + i];
        s += v; s2 += v * v;
    }
    __shared__ float sh0[256];
    __shared__ float sh1[256];
    sh0[threadIdx.x] = s; sh1[threadIdx.x] = s2;
    __syncthreads();
    for (int off = blockDim.x >> 1; off > 0; off >>= 1) {
        if ((int)threadIdx.x < off) {
            sh0[threadIdx.x] += sh0[threadIdx.x + off];
            sh1[threadIdx.x] += sh1[threadIdx.x + off];
        }
        __syncthreads();
    }
    if (threadIdx.x == 0) {
        atomicAdd(&acc[accOff + b],     sh0[0]);
        atomicAdd(&acc[accOff + 4 + b], sh1[0]);
    }
}

__global__ void finalize_norm_kernel(float* __restrict__ acc, int B, int perImage)
{
    if (blockIdx.x == 0 && threadIdx.x == 0) {
        const float n = (float)perImage;
        float sumAll = 0.0f, var1 = 0.0f, var2 = 0.0f;
        for (int b = 0; b < B; ++b) {
            sumAll += acc[b] + acc[8 + b];
            const float m1 = acc[b] / n;
            const float m2 = acc[8 + b] / n;
            var1 += acc[4 + b]  / n - m1 * m1;
            var2 += acc[12 + b] / n - m2 * m2;
        }
        const float mean = 0.5f * sumAll / (n * (float)B);
        const float var  = 0.5f * (var1 + var2) / (float)B;
        acc[16] = mean;
        acc[17] = 1.0f / sqrtf(var + 1e-16f);
    }
}

// ---------------------------------------------------------------------------
// 81-tap cost volume on normalized features, leaky-relu, into concat channels.
// ---------------------------------------------------------------------------
__global__ void costvol_kernel(const float* __restrict__ f1,
                               const float* __restrict__ w2,
                               const float* __restrict__ norm,
                               float* __restrict__ xin, int xinStride, int cvOff,
                               int B, int H, int W)
{
    const int P = B * H * W;
    const int p = blockIdx.x * blockDim.x + threadIdx.x;
    if (p >= P) return;
    const int x = p % W;
    int t = p / W;
    const int y = t % H;
    const int b = t / H;

    const float mean = norm[0], inv = norm[1];
    float a[32];
    #pragma unroll
    for (int c = 0; c < 32; ++c)
        a[c] = (f1[(size_t)p * 32 + c] - mean) * inv;

    for (int i = 0; i < 9; ++i) {
        for (int j = 0; j < 9; ++j) {
            const int yy = y + i - 4, xx = x + j - 4;
            float s = 0.0f;
            if (yy >= 0 && yy < H && xx >= 0 && xx < W) {
                const float* q = &w2[(((size_t)b * H + yy) * W + xx) * 32];
                #pragma unroll
                for (int c = 0; c < 32; ++c)
                    s += a[c] * ((q[c] - mean) * inv);
            }
            float cv = s * (1.0f / 32.0f);
            cv = cv > 0.0f ? cv : 0.1f * cv;
            xin[(size_t)p * xinStride + cvOff + i * 9 + j] = cv;
        }
    }
}

// Strided channel copy: dst[p*dstStride + dstOff + c] = src[p*srcStride + c]
__global__ void copy_ch_kernel(const float* __restrict__ src, int srcStride,
                               float* __restrict__ dst, int dstStride, int dstOff,
                               int C, int P)
{
    const int i = blockIdx.x * blockDim.x + threadIdx.x;
    if (i >= P * C) return;
    const int p = i / C, c = i - p * C;
    dst[(size_t)p * dstStride + dstOff + c] = src[(size_t)p * srcStride + c];
}

// Bilinear 2x upsample of a 2-channel flow field (half-pixel centers), *2.
__global__ void upflow_kernel(const float* __restrict__ src, int B, int H, int W,
                              float* __restrict__ dst)
{
    const int Ho = 2 * H, Wo = 2 * W;
    const int P = B * Ho * Wo;
    const int i = blockIdx.x * blockDim.x + threadIdx.x;
    if (i >= P) return;
    const int x = i % Wo;
    int t = i / Wo;
    const int y = t % Ho;
    const int b = t / Ho;

    const float sy = (y + 0.5f) * 0.5f - 0.5f;
    const float sx = (x + 0.5f) * 0.5f - 0.5f;
    const float y0f = floorf(sy), x0f = floorf(sx);
    const float wy = sy - y0f, wx = sx - x0f;
    const int y0 = (int)y0f, x0 = (int)x0f;
    const int y0c = iclamp(y0, 0, H - 1),     y1c = iclamp(y0 + 1, 0, H - 1);
    const int x0c = iclamp(x0, 0, W - 1),     x1c = iclamp(x0 + 1, 0, W - 1);

    const float* s = src + (size_t)b * H * W * 2;
    #pragma unroll
    for (int c = 0; c < 2; ++c) {
        const float v00 = s[((size_t)y0c * W + x0c) * 2 + c];
        const float v01 = s[((size_t)y0c * W + x1c) * 2 + c];
        const float v10 = s[((size_t)y1c * W + x0c) * 2 + c];
        const float v11 = s[((size_t)y1c * W + x1c) * 2 + c];
        const float v = (1.0f - wy) * ((1.0f - wx) * v00 + wx * v01) +
                        wy          * ((1.0f - wx) * v10 + wx * v11);
        dst[(size_t)i * 2 + c] = 2.0f * v;
    }
}

// 4x4 stride-2 transposed conv (context upsampler, 32->32, SAME).
__global__ void convt_kernel(const float* __restrict__ in, int inStride, int inOff,
                             const float* __restrict__ w,   // [4][4][32][32]
                             const float* __restrict__ bias,
                             float* __restrict__ dst,        // [B,2H,2W,32]
                             int B, int Hin, int Win)
{
    const int Ho = 2 * Hin, Wo = 2 * Win;
    const int N = B * Ho * Wo * 32;
    const int idx = blockIdx.x * blockDim.x + threadIdx.x;
    if (idx >= N) return;
    const int co = idx & 31;
    int p = idx >> 5;
    const int x = p % Wo;
    int t = p / Wo;
    const int y = t % Ho;
    const int b = t / Ho;

    float s = bias[co];
    for (int ky = 0; ky < 4; ++ky) {
        const int ty = y + 1 - ky;
        if (ty < 0 || (ty & 1)) continue;
        const int iy = ty >> 1;
        if (iy >= Hin) continue;
        for (int kx = 0; kx < 4; ++kx) {
            const int tx = x + 1 - kx;
            if (tx < 0 || (tx & 1)) continue;
            const int ix = tx >> 1;
            if (ix >= Win) continue;
            const float* ip = &in[(((size_t)b * Hin + iy) * Win + ix) * inStride + inOff];
            const float* wp = &w[((size_t)(ky * 4 + kx) * 32) * 32 + co];
            #pragma unroll
            for (int ci = 0; ci < 32; ++ci)
                s += ip[ci] * wp[(size_t)ci * 32];
        }
    }
    dst[idx] = s;
}

__global__ void add_kernel(float* __restrict__ dst, const float* __restrict__ src, int N)
{
    const int i = blockIdx.x * blockDim.x + threadIdx.x;
    if (i < N) dst[i] += src[i];
}

__global__ void zero_kernel(float* __restrict__ p, int n)
{
    const int i = blockIdx.x * blockDim.x + threadIdx.x;
    if (i < n) p[i] = 0.0f;
}

// ---------------------------------------------------------------------------
// Host-side orchestration
// ---------------------------------------------------------------------------
static inline int pad32(int v) { return (v + 31) & ~31; }
static inline int pad16(int v) { return (v + 15) & ~15; }

static void launch_conv(hipStream_t s,
                        const float* in, int inStride, int inOff, int Cin,
                        const _Float16* wpack, int CinPad, const float* b,
                        float* out, int outStride, int outOff, int Cout,
                        const float* resid, int residStride,
                        int B, int H, int W, int dil, int leaky)
{
    const int coutPad = pad16(Cout);
    const int nt = (coutPad % 64 == 0) ? 4 : (coutPad % 32 == 0) ? 2 : 1;
    const long long P = (long long)B * H * W;
    const long long waves = ((P + 15) / 16) * (coutPad / (16 * nt));
    const int blocks = (int)((waves + 3) / 4);
    if (nt == 4)
        conv3x3_wmma_kernel<4><<<blocks, 128, 0, s>>>(in, inStride, inOff, Cin,
            wpack, CinPad, b, out, outStride, outOff, Cout, resid, residStride,
            B, H, W, dil, leaky);
    else if (nt == 2)
        conv3x3_wmma_kernel<2><<<blocks, 128, 0, s>>>(in, inStride, inOff, Cin,
            wpack, CinPad, b, out, outStride, outOff, Cout, resid, residStride,
            B, H, W, dil, leaky);
    else
        conv3x3_wmma_kernel<1><<<blocks, 128, 0, s>>>(in, inStride, inOff, Cin,
            wpack, CinPad, b, out, outStride, outOff, Cout, resid, residStride,
            B, H, W, dil, leaky);
}

extern "C" void kernel_launch(void* const* d_in, const int* in_sizes, int n_in,
                              void* d_out, int out_size, void* d_ws, size_t ws_size,
                              hipStream_t stream)
{
    (void)in_sizes; (void)n_in; (void)out_size; (void)ws_size;
    const int B = 4;
    static const int LVL_H[5] = {256, 128, 64, 32, 16};
    static const int LVL_W[5] = {320, 160, 80, 40, 20};

    // ---- inputs: interleaved features1_l{l}, features2_l{l} ----
    const float* f1[5]; const float* f2[5];
    for (int l = 0; l < 5; ++l) {
        f1[l] = (const float*)d_in[2 * l];
        f2[l] = (const float*)d_in[2 * l + 1];
    }
    // ---- params (jax pytree flatten order) ----
    int pi = 10;
    const float *ctxw[5], *ctxb[5];
    for (int l = 1; l <= 4; ++l) { ctxw[l] = (const float*)d_in[pi++]; ctxb[l] = (const float*)d_in[pi++]; }
    const float *cw[5][5], *cb[5][5], *fhb[5], *fhw[5];
    for (int l = 1; l <= 4; ++l) {
        for (int i = 0; i < 5; ++i) { cw[l][i] = (const float*)d_in[pi++]; cb[l][i] = (const float*)d_in[pi++]; }
        fhb[l] = (const float*)d_in[pi++];   // 'flow_b' sorts before 'flow_w'
        fhw[l] = (const float*)d_in[pi++];
    }
    const float *raw[6], *rab[6], *rbw[6], *rbb[6];
    for (int i = 0; i < 6; ++i) { raw[i] = (const float*)d_in[pi++]; rab[i] = (const float*)d_in[pi++]; }
    for (int i = 0; i < 6; ++i) { rbw[i] = (const float*)d_in[pi++]; rbb[i] = (const float*)d_in[pi++]; }

    float* out = (float*)d_out;
    float* outL[5];
    outL[1] = out;                               // [4,128,160,2]
    outL[2] = out + 163840;                      // [4, 64, 80,2]
    outL[3] = out + 163840 + 40960;              // [4, 32, 40,2]
    outL[4] = out + 163840 + 40960 + 10240;      // [4, 16, 20,2]

    // ---- workspace layout (floats; every region 128B aligned) ----
    const int XS = 608;                          // channel stride of concat tensor
    float* XIN = (float*)d_ws;                                   // 4*128*160*608
    float* W2  = XIN + (size_t)4 * 128 * 160 * 608;              // warped2, 32ch
    float* FUP = W2  + (size_t)4 * 128 * 160 * 32;               // flow_up, 2ch
    float* CUP = FUP + (size_t)4 * 128 * 160 * 2;                // ctx_up, 32ch
    float* ACC = CUP + (size_t)4 * 128 * 160 * 32;               // 32 floats
    float* RA  = ACC + 32;                                       // refine tmp 128ch
    float* RB  = RA  + (size_t)4 * 128 * 160 * 128;              // refine tmp 2ch
    _Float16* packCur = (_Float16*)(RB + (size_t)4 * 128 * 160 * 2);

    // ---- pre-pack all conv weights into f16 B-fragment order ----
    struct Pk { const _Float16* p; int cinPad; };
    auto packw = [&](const float* w, int Cin, int Cout) -> Pk {
        const int cinPad  = pad32(Cin);
        const int coutPad = pad16(Cout);
        const int N = coutPad * 9 * cinPad;
        pack_weights_kernel<<<(N + 255) / 256, 256, 0, stream>>>(
            w, Cin, Cout, packCur, cinPad, coutPad);
        Pk r{packCur, cinPad};
        packCur += (size_t)N;
        return r;
    };

    static const int CL[5] = {128, 128, 96, 64, 32};
    Pk cpk[5][5], fhpk[5], rapk[6], rbpk[6];
    for (int l = 4; l >= 1; --l) {
        int ch = (l == 4) ? 113 : 147;
        for (int i = 0; i < 5; ++i) { cpk[l][i] = packw(cw[l][i], ch, CL[i]); ch += CL[i]; }
        fhpk[l] = packw(fhw[l], 32, 2);
    }
    static const int RC[6] = {128, 128, 128, 96, 64, 32};
    static const int RD[6] = {1, 2, 4, 8, 16, 1};
    for (int i = 0; i < 6; ++i) {
        rapk[i] = packw(raw[i], i == 0 ? 34 : 2, RC[i]);
        rbpk[i] = packw(rbw[i], RC[i], 2);
    }

    // ---- coarse-to-fine pyramid ----
    bool first = true;
    for (int l = 4; l >= 1; --l) {
        const int H = LVL_H[l], W = LVL_W[l], P = B * H * W;

        const float* w2src;
        if (first) {
            w2src = f2[l];
        } else {
            warp_kernel<<<(P + 127) / 128, 128, 0, stream>>>(f2[l], FUP, W2, B, H, W, 32);
            w2src = W2;
        }

        zero_kernel<<<1, 32, 0, stream>>>(ACC, 32);
        moments_kernel<<<dim3(32, B), 256, 0, stream>>>(f1[l], H * W * 32, ACC, 0);
        moments_kernel<<<dim3(32, B), 256, 0, stream>>>(w2src, H * W * 32, ACC, 8);
        finalize_norm_kernel<<<1, 1, 0, stream>>>(ACC, B, H * W * 32);

        int cvOff, featOff, ch;
        if (first) { cvOff = 0; featOff = 81; ch = 113; }
        else {
            // concat order: [context_up(32), flow_up(2), cv(81), features1(32)]
            copy_ch_kernel<<<(P * 32 + 255) / 256, 256, 0, stream>>>(CUP, 32, XIN, XS, 0, 32, P);
            copy_ch_kernel<<<(P * 2  + 255) / 256, 256, 0, stream>>>(FUP, 2,  XIN, XS, 32, 2, P);
            cvOff = 34; featOff = 115; ch = 147;
        }
        costvol_kernel<<<(P + 127) / 128, 128, 0, stream>>>(f1[l], w2src, ACC + 16,
                                                            XIN, XS, cvOff, B, H, W);
        copy_ch_kernel<<<(P * 32 + 255) / 256, 256, 0, stream>>>(f1[l], 32, XIN, XS, featOff, 32, P);

        // dense-net decoder block: outputs appended into the concat tensor
        for (int i = 0; i < 5; ++i) {
            launch_conv(stream, XIN, XS, 0, ch, cpk[l][i].p, cpk[l][i].cinPad, cb[l][i],
                        XIN, XS, ch, CL[i], nullptr, 0, B, H, W, 1, 1);
            ch += CL[i];
        }
        const int ctxOff = ch - 32;   // context = last block output

        // flow head (+ flow_up accumulate), written straight into d_out slice
        launch_conv(stream, XIN, XS, ctxOff, 32, fhpk[l].p, fhpk[l].cinPad, fhb[l],
                    outL[l], 2, 0, 2, first ? nullptr : FUP, 2, B, H, W, 1, 0);

        if (l > 1) {
            upflow_kernel<<<(B * 4 * H * W + 127) / 128, 128, 0, stream>>>(outL[l], B, H, W, FUP);
            convt_kernel<<<(B * 4 * H * W * 32 + 255) / 256, 256, 0, stream>>>(
                XIN, XS, ctxOff, ctxw[l], ctxb[l], CUP, B, H, W);
        }
        first = false;
    }

    // ---- refinement at level 1: in = concat(context(32), flow(2)) ----
    {
        const int H = 128, W = 160, P = B * H * W;
        const int ctxOff = 563;                       // level-1 context in XIN
        copy_ch_kernel<<<(P * 2 + 255) / 256, 256, 0, stream>>>(outL[1], 2, XIN, XS, ctxOff + 32, 2, P);

        for (int i = 0; i < 6; ++i) {
            const float* src; int ss, so, cin;
            if (i == 0) { src = XIN; ss = XS; so = ctxOff; cin = 34; }
            else        { src = RB;  ss = 2;  so = 0;      cin = 2;  }
            launch_conv(stream, src, ss, so, cin, rapk[i].p, rapk[i].cinPad, rab[i],
                        RA, 128, 0, RC[i], nullptr, 0, B, H, W, RD[i], 1);
            launch_conv(stream, RA, 128, 0, RC[i], rbpk[i].p, rbpk[i].cinPad, rbb[i],
                        RB, 2, 0, 2, nullptr, 0, B, H, W, 1, 0);
        }
        add_kernel<<<(P * 2 + 255) / 256, 256, 0, stream>>>(outL[1], RB, P * 2);
    }
}
